// LowMemLMDViT_49443663511626
// MI455X (gfx1250) — compile-verified, matrix operand order checked by reading, not run on hardware
//
#include <hip/hip_runtime.h>
#include <hip/hip_bf16.h>

// ---------------- types ----------------
typedef _Float16 half8  __attribute__((ext_vector_type(8)));
typedef _Float16 half16 __attribute__((ext_vector_type(16)));
typedef float    float8 __attribute__((ext_vector_type(8)));

// ---------------- problem constants ----------------
#define BATCH  32
#define HDIM   56
#define WDIM   56
#define CC     384
#define NHEADS 12
#define WSZ    7
#define SHIFT  3
#define HIDDEN 768
#define HDHEAD 32
#define NTOK   49          // WSZ*WSZ
#define NWIN   64          // (56/7)^2
#define BWIN   2048        // BATCH*NWIN
#define MTOK   100352      // BWIN*NTOK = BATCH*HDIM*WDIM
#define QSCALE 0.17677669529663687f
#define LNEPS  1e-5f
#define NEGBIG -1e30f

// ---------------- workspace layout (bytes) ----------------
static constexpr size_t OFF_QKVW = 0;                               // 1152*384*2
static constexpr size_t OFF_PRJW = OFF_QKVW + (size_t)1152*384*2;   // 384*384*2
static constexpr size_t OFF_FC1W = OFF_PRJW + (size_t)384*384*2;    // 768*384*2
static constexpr size_t OFF_FC2W = OFF_FC1W + (size_t)768*384*2;    // 384*768*2
static constexpr size_t OFF_BIAS = OFF_FC2W + (size_t)384*768*2;    // 64*12*64*64*4 (padded)
static constexpr size_t OFF_XA   = OFF_BIAS + (size_t)NWIN*NHEADS*64*64*4;     // xw / attn_out
static constexpr size_t OFF_Q    = OFF_XA + (size_t)MTOK*CC*2;
static constexpr size_t OFF_K    = OFF_Q  + (size_t)MTOK*CC*2;
static constexpr size_t OFF_V    = OFF_K  + (size_t)MTOK*CC*2;
static constexpr size_t QKVSZ    = (size_t)MTOK*CC;                 // elements per q/k/v tensor
// h (MTOK x 768 f16) aliases [OFF_Q, OFF_V);  xm (MTOK x 384 f16) aliases OFF_V.

// ---------------- helpers ----------------
__device__ inline float8 wmma_f16(half16 a, half16 b, float8 c) {
  // v_wmma_f32_16x16x32_f16  (neg_a, A, neg_b, B, c_mod, C, reuse_a, reuse_b)
  return __builtin_amdgcn_wmma_f32_16x16x32_f16(false, a, false, b, (short)0, c, false, false);
}

// Load one 16x16x32-f16 A- or B-fragment row: lane L holds row (L&15),
// halfs [kk + kh .. kk+kh+7] and [kk+16+kh .. kk+16+kh+7], kh = (L>>4)*8.
__device__ inline half16 load_frag(const _Float16* base, int row, int ld, int kk, int lane) {
  const int kh = (lane >> 4) << 3;
  const _Float16* p = base + (size_t)row * ld + kk + kh;
  half8 lo = *(const half8*)(p);
  half8 hi = *(const half8*)(p + 16);
  return __builtin_shufflevector(lo, hi, 0,1,2,3,4,5,6,7,8,9,10,11,12,13,14,15);
}

__device__ inline float wave_red_sum(float v) {
  #pragma unroll
  for (int d = 16; d; d >>= 1) v += __shfl_xor(v, d, 32);
  return v;
}
__device__ inline float half_red_max(float v) {   // within 16-lane groups (wave32 halves)
  #pragma unroll
  for (int d = 8; d; d >>= 1) v = fmaxf(v, __shfl_xor(v, d, 32));
  return v;
}
__device__ inline float half_red_sum(float v) {
  #pragma unroll
  for (int d = 8; d; d >>= 1) v += __shfl_xor(v, d, 32);
  return v;
}

// ---------------- kernel 0a: f32 -> f16 weight convert ----------------
__global__ void swin_cvt_f16(const float* __restrict__ s, _Float16* __restrict__ d, int n) {
  int i = blockIdx.x * 256 + threadIdx.x;
  if (i < n) d[i] = (_Float16)s[i];
}

// ---------------- kernel 0b: fused rel-bias + shift-mask table, PADDED to 64x64 ----------------
// out[((nw*12+h)*64+r)*64+c]:
//   r<49 && c<49 : rpb[rel_idx(r,c)*12+h] + mask(nw,r,c)
//   c>=49        : -1e30   (kills padded key columns in softmax)
//   r>=49, c<49  : 0       (padded query rows: finite, discarded at store)
__global__ void swin_build_bias(const float* __restrict__ rpb, float* __restrict__ out) {
  int idx = blockIdx.x * 256 + threadIdx.x;
  if (idx >= NWIN * NHEADS * 64 * 64) return;
  int c  = idx & 63;
  int r  = (idx >> 6) & 63;
  int h  = (idx >> 12) % NHEADS;
  int nw = idx / (NHEADS * 64 * 64);
  float v;
  if (c >= NTOK) {
    v = NEGBIG;
  } else if (r >= NTOK) {
    v = 0.0f;
  } else {
    int yr = r / WSZ, xr = r % WSZ, yc = c / WSZ, xc = c % WSZ;
    int rel = (yr - yc + WSZ - 1) * (2 * WSZ - 1) + (xr - xc + WSZ - 1);
    v = rpb[rel * NHEADS + h];
    int wh = nw >> 3, ww = nw & 7;
    auto reg = [](int p) { return (p < HDIM - WSZ) ? 0 : ((p < HDIM - SHIFT) ? 1 : 2); };
    int idr = reg(wh * WSZ + yr) * 3 + reg(ww * WSZ + xr);
    int idc = reg(wh * WSZ + yc) * 3 + reg(ww * WSZ + xc);
    if (idr != idc) v -= 100.0f;
  }
  out[idx] = v;
}

// ---------------- kernel 1: LN1 + roll(-3,-3) + window partition -> f16 ----------------
// one wave per destination (window-ordered) token
__global__ void __launch_bounds__(256)
swin_ln1_window(const float* __restrict__ x, const float* __restrict__ w,
                const float* __restrict__ b, _Float16* __restrict__ xw) {
  const int lane = threadIdx.x & 31, wv = threadIdx.x >> 5;
  const int t = blockIdx.x * 8 + wv;                 // [0, MTOK)
  const int b_ = t / NTOK, nt = t % NTOK;
  const int bb = b_ >> 6, widx = b_ & 63;
  const int wh = widx >> 3, ww = widx & 7;
  const int ti = nt / WSZ, tj = nt % WSZ;
  const int y  = (wh * WSZ + ti + SHIFT) % HDIM;     // roll by -SHIFT
  const int xc = (ww * WSZ + tj + SHIFT) % WDIM;
  const float* src = x + ((size_t)bb * (HDIM * WDIM) + y * WDIM + xc) * CC;

  float vals[12]; float s = 0.f;
  #pragma unroll
  for (int i = 0; i < 12; ++i) { vals[i] = src[lane + 32 * i]; s += vals[i]; }
  s = wave_red_sum(s);
  const float mu = s * (1.0f / CC);
  float s2 = 0.f;
  #pragma unroll
  for (int i = 0; i < 12; ++i) { float d = vals[i] - mu; s2 += d * d; }
  s2 = wave_red_sum(s2);
  const float rs = rsqrtf(s2 * (1.0f / CC) + LNEPS);
  _Float16* dst = xw + (size_t)t * CC;
  #pragma unroll
  for (int i = 0; i < 12; ++i) {
    int c = lane + 32 * i;
    dst[c] = (_Float16)((vals[i] - mu) * rs * w[c] + b[c]);
  }
}

// ---------------- kernel 5: LN2 -> f16 (linear tokens) ----------------
__global__ void __launch_bounds__(256)
swin_ln2(const float* __restrict__ x, const float* __restrict__ w,
         const float* __restrict__ b, _Float16* __restrict__ xm) {
  const int lane = threadIdx.x & 31, wv = threadIdx.x >> 5;
  const int t = blockIdx.x * 8 + wv;
  const float* src = x + (size_t)t * CC;
  float vals[12]; float s = 0.f;
  #pragma unroll
  for (int i = 0; i < 12; ++i) { vals[i] = src[lane + 32 * i]; s += vals[i]; }
  s = wave_red_sum(s);
  const float mu = s * (1.0f / CC);
  float s2 = 0.f;
  #pragma unroll
  for (int i = 0; i < 12; ++i) { float d = vals[i] - mu; s2 += d * d; }
  s2 = wave_red_sum(s2);
  const float rs = rsqrtf(s2 * (1.0f / CC) + LNEPS);
  _Float16* dst = xm + (size_t)t * CC;
  #pragma unroll
  for (int i = 0; i < 12; ++i) {
    int c = lane + 32 * i;
    dst[c] = (_Float16)((vals[i] - mu) * rs * w[c] + b[c]);
  }
}

// ---------------- generic WMMA GEMM:  out = A(MxK,f16) @ W(NxK,f16)^T + bias ----------------
// Block = 256 (8 waves). Wave tile = 32(M) x 64(N): 2 A-frags x 4 B-frags = 8 WMMA / K-step.
// grid = (M/256, N/64).
// MODE 0: QKV scatter to q/k/v (q scaled);  MODE 1: proj + window-reverse + roll + residual;
// MODE 2: FC1 + exact GELU -> f16;          MODE 3: FC2 + residual accumulate into f32 out.
template <int MODE>
__global__ void __launch_bounds__(256)
swin_gemm_wmma(const _Float16* __restrict__ A, const _Float16* __restrict__ W,
               const float* __restrict__ bias, int K,
               _Float16* __restrict__ o16, float* __restrict__ o32,
               const float* __restrict__ add32) {
  const int lane  = threadIdx.x & 31;
  const int wv    = threadIdx.x >> 5;
  const int mbase = blockIdx.x * 256 + wv * 32;
  const int nbase = blockIdx.y * 64;
  const int mr0   = mbase + (lane & 15);
  const int mr1   = mr0 + 16;

  float8 acc[2][4] = {};
  for (int kk = 0; kk < K; kk += 32) {
    if (kk + 32 < K) {
      __builtin_prefetch(A + (size_t)mr0 * K + kk + 32, 0, 1);   // global_prefetch_b8
      __builtin_prefetch(A + (size_t)mr1 * K + kk + 32, 0, 1);
    }
    half16 a0 = load_frag(A, mr0, K, kk, lane);
    half16 a1 = load_frag(A, mr1, K, kk, lane);
    #pragma unroll
    for (int t = 0; t < 4; ++t) {
      half16 bf = load_frag(W, nbase + t * 16 + (lane & 15), K, kk, lane);
      acc[0][t] = wmma_f16(a0, bf, acc[0][t]);
      acc[1][t] = wmma_f16(a1, bf, acc[1][t]);
    }
  }

  const int hi = lane >> 4;
  #pragma unroll
  for (int m2 = 0; m2 < 2; ++m2) {
    #pragma unroll
    for (int t = 0; t < 4; ++t) {
      const int n  = nbase + t * 16 + (lane & 15);
      const float bv = bias[n];
      #pragma unroll
      for (int j = 0; j < 8; ++j) {
        const int row = mbase + m2 * 16 + j + hi * 8;
        float v = acc[m2][t][j] + bv;
        if constexpr (MODE == 0) {            // QKV scatter
          int s    = n / CC;
          int rem  = n - s * CC;
          int head = rem >> 5;
          int d    = rem & 31;
          if (s == 0) v *= QSCALE;
          int b_ = row / NTOK, nt = row - b_ * NTOK;
          size_t dst = (((size_t)b_ * NHEADS + head) * NTOK + nt) * HDHEAD + d;
          o16[(size_t)s * QKVSZ + dst] = (_Float16)v;
        } else if constexpr (MODE == 1) {     // proj + window reverse + roll(+3) + residual
          int b_ = row / NTOK, nt = row - b_ * NTOK;
          int bb = b_ >> 6, widx = b_ & 63;
          int wh = widx >> 3, ww = widx & 7;
          int ti = nt / WSZ, tj = nt % WSZ;
          int y  = (wh * WSZ + ti + SHIFT) % HDIM;
          int xc = (ww * WSZ + tj + SHIFT) % WDIM;
          size_t pix = ((size_t)bb * (HDIM * WDIM) + y * WDIM + xc) * CC + n;
          o32[pix] = v + add32[pix];
        } else if constexpr (MODE == 2) {     // FC1 + exact GELU
          float g = 0.5f * v * (1.0f + erff(v * 0.70710678118654752f));
          o16[(size_t)row * HIDDEN + n] = (_Float16)g;
        } else {                              // FC2 + residual
          size_t idx = (size_t)row * CC + n;
          o32[idx] = o32[idx] + v;
        }
      }
    }
  }
}

// ---------------- kernel 3: windowed attention (one wave per (window, head)) ----------------
__global__ void __launch_bounds__(128)
swin_attention(const _Float16* __restrict__ q, const _Float16* __restrict__ k,
               const _Float16* __restrict__ v, const float* __restrict__ bias_comb,
               _Float16* __restrict__ attn_out) {
  __shared__ _Float16 lds_p[4][64 * 64];    // softmax probs, row-major 64x64
  __shared__ _Float16 lds_vt[4][32 * 64];   // V transposed: vt[c][r]

  const int lane = threadIdx.x & 31, wv = threadIdx.x >> 5;
  const int w    = blockIdx.x / 3;
  const int head = (blockIdx.x % 3) * 4 + wv;
  const size_t base = ((size_t)w * NHEADS + head) * NTOK * HDHEAD;
  const _Float16* qp = q + base;
  const _Float16* kp = k + base;
  const _Float16* vp = v + base;
  _Float16* P  = lds_p[wv];
  _Float16* VT = lds_vt[wv];

  // stage V transposed (pad rows >= 49 with zero)
  for (int i = lane; i < 64 * HDHEAD; i += 32) {
    int r = i >> 5, c = i & 31;
    VT[c * 64 + r] = (r < NTOK) ? vp[r * HDHEAD + c] : (_Float16)0.0f;
  }

  // B-fragments of K (K^T columns are K rows -> contiguous loads)
  const half16 hz = {};
  half16 bf[4];
  #pragma unroll
  for (int t = 0; t < 4; ++t) {
    int n = t * 16 + (lane & 15);
    bf[t] = (n < NTOK) ? load_frag(kp, n, HDHEAD, 0, lane) : hz;
  }

  // padded 64x64 fused bias+mask: unguarded coalesced loads
  const float* bw = bias_comb + ((size_t)(w & 63) * NHEADS + head) * 64 * 64;
  const int hi = lane >> 4;

  // phase 1: scores + fused bias/mask + row softmax, write P to LDS
  for (int mi = 0; mi < 4; ++mi) {
    int m = mi * 16 + (lane & 15);
    half16 af = (m < NTOK) ? load_frag(qp, m, HDHEAD, 0, lane) : hz;
    float8 s[4];
    #pragma unroll
    for (int t = 0; t < 4; ++t) { float8 z = {}; s[t] = wmma_f16(af, bf[t], z); }

    float vals[4][8];
    #pragma unroll
    for (int t = 0; t < 4; ++t) {
      const int c = t * 16 + (lane & 15);
      #pragma unroll
      for (int j = 0; j < 8; ++j) {
        const int r = mi * 16 + j + hi * 8;
        vals[t][j] = s[t][j] + bw[r * 64 + c];
      }
    }

    #pragma unroll
    for (int j = 0; j < 8; ++j) {
      float mx = fmaxf(fmaxf(vals[0][j], vals[1][j]), fmaxf(vals[2][j], vals[3][j]));
      mx = half_red_max(mx);
      float sm = 0.f;
      #pragma unroll
      for (int t = 0; t < 4; ++t) { vals[t][j] = __expf(vals[t][j] - mx); sm += vals[t][j]; }
      sm = half_red_sum(sm);
      float inv = 1.0f / sm;
      int r = mi * 16 + j + hi * 8;
      #pragma unroll
      for (int t = 0; t < 4; ++t) {
        int c = t * 16 + (lane & 15);
        P[r * 64 + c] = (_Float16)(vals[t][j] * inv);
      }
    }
  }

  __syncthreads();   // LDS visibility (P, VT) across lanes

  // phase 2: O = P(64x64) @ V(64x32), fragments from LDS (ds_load_b128)
  for (int mi = 0; mi < 4; ++mi) {
    int m = mi * 16 + (lane & 15);
    half16 a0 = load_frag(P, m, 64, 0, lane);
    half16 a1 = load_frag(P, m, 64, 32, lane);
    #pragma unroll
    for (int ni = 0; ni < 2; ++ni) {
      int n = ni * 16 + (lane & 15);
      half16 b0 = load_frag(VT, n, 64, 0, lane);
      half16 b1 = load_frag(VT, n, 64, 32, lane);
      float8 o = {};
      o = wmma_f16(a0, b0, o);
      o = wmma_f16(a1, b1, o);
      #pragma unroll
      for (int j = 0; j < 8; ++j) {
        int r = mi * 16 + j + hi * 8;
        if (r < NTOK) {
          size_t dst = ((size_t)w * NTOK + r) * CC + head * HDHEAD + ni * 16 + (lane & 15);
          attn_out[dst] = (_Float16)(o[j]);
        }
      }
    }
  }
}

// ---------------- host launcher ----------------
extern "C" void kernel_launch(void* const* d_in, const int* in_sizes, int n_in,
                              void* d_out, int out_size, void* d_ws, size_t ws_size,
                              hipStream_t stream) {
  const float* x       = (const float*)d_in[0];
  const float* n1w     = (const float*)d_in[1];
  const float* n1b     = (const float*)d_in[2];
  const float* qkv_w   = (const float*)d_in[3];
  const float* qkv_b   = (const float*)d_in[4];
  const float* rpb     = (const float*)d_in[5];
  const float* proj_w  = (const float*)d_in[6];
  const float* proj_b  = (const float*)d_in[7];
  const float* n2w     = (const float*)d_in[8];
  const float* n2b     = (const float*)d_in[9];
  const float* fc1_w   = (const float*)d_in[10];
  const float* fc1_b   = (const float*)d_in[11];
  const float* fc2_w   = (const float*)d_in[12];
  const float* fc2_b   = (const float*)d_in[13];
  float* out = (float*)d_out;

  char* ws = (char*)d_ws;
  _Float16* qkvw16 = (_Float16*)(ws + OFF_QKVW);
  _Float16* prjw16 = (_Float16*)(ws + OFF_PRJW);
  _Float16* fc1w16 = (_Float16*)(ws + OFF_FC1W);
  _Float16* fc2w16 = (_Float16*)(ws + OFF_FC2W);
  float*    biasc  = (float*)   (ws + OFF_BIAS);
  _Float16* xw     = (_Float16*)(ws + OFF_XA);   // reused as attn_out
  _Float16* qbuf   = (_Float16*)(ws + OFF_Q);
  _Float16* kbuf   = (_Float16*)(ws + OFF_K);
  _Float16* vbuf   = (_Float16*)(ws + OFF_V);
  _Float16* hbuf   = (_Float16*)(ws + OFF_Q);    // aliases q+k (dead after attention)
  _Float16* xmbuf  = (_Float16*)(ws + OFF_V);    // aliases v   (dead after attention)
  _Float16* attn_o = xw;                         // aliases xw  (dead after QKV)

  auto cdiv = [](int a, int b) { return (a + b - 1) / b; };

  // 0) weights -> f16, fused padded bias+mask table
  swin_cvt_f16<<<cdiv(1152 * 384, 256), 256, 0, stream>>>(qkv_w,  qkvw16, 1152 * 384);
  swin_cvt_f16<<<cdiv(384 * 384,  256), 256, 0, stream>>>(proj_w, prjw16, 384 * 384);
  swin_cvt_f16<<<cdiv(768 * 384,  256), 256, 0, stream>>>(fc1_w,  fc1w16, 768 * 384);
  swin_cvt_f16<<<cdiv(384 * 768,  256), 256, 0, stream>>>(fc2_w,  fc2w16, 384 * 768);
  swin_build_bias<<<cdiv(NWIN * NHEADS * 64 * 64, 256), 256, 0, stream>>>(rpb, biasc);

  // 1) LN1 + roll + window partition (f16)
  swin_ln1_window<<<MTOK / 8, 256, 0, stream>>>(x, n1w, n1b, xw);

  // 2) QKV GEMM (100352 x 1152, K=384) with q/k/v scatter
  swin_gemm_wmma<0><<<dim3(MTOK / 256, 1152 / 64), 256, 0, stream>>>(
      xw, qkvw16, qkv_b, CC, qbuf, nullptr, nullptr);

  // 3) windowed attention
  swin_attention<<<BWIN * 3, 128, 0, stream>>>(qbuf, kbuf, vbuf, biasc, attn_o);

  // 4) proj GEMM + window reverse + roll + residual -> d_out (x2, fp32)
  swin_gemm_wmma<1><<<dim3(MTOK / 256, 384 / 64), 256, 0, stream>>>(
      attn_o, prjw16, proj_b, CC, nullptr, out, x);

  // 5) LN2 (f16)
  swin_ln2<<<MTOK / 8, 256, 0, stream>>>(out, n2w, n2b, xmbuf);

  // 6) FC1 GEMM + GELU (f16)
  swin_gemm_wmma<2><<<dim3(MTOK / 256, HIDDEN / 64), 256, 0, stream>>>(
      xmbuf, fc1w16, fc1_b, CC, hbuf, nullptr, nullptr);

  // 7) FC2 GEMM + residual accumulate into d_out
  swin_gemm_wmma<3><<<dim3(MTOK / 256, 384 / 64), 256, 0, stream>>>(
      hbuf, fc2w16, fc2_b, HIDDEN, nullptr, out, nullptr);
}
